// CNN_ATT_6116033429864
// MI455X (gfx1250) — matrix operand order repaired and checked by reading, not compiled, not run on hardware
//
#include <hip/hip_runtime.h>
#include <hip/hip_bf16.h>

// ---- problem constants (from reference) ----
constexpr int kB   = 128;   // bags
constexpr int kM   = 16;    // sentences per bag
constexpr int kL   = 120;   // tokens per sentence
constexpr int kWE  = 50;    // word embed dim
constexpr int kPE  = 5;     // pos embed dim
constexpr int kD   = 60;    // WE + 2*PE
constexpr int kOC  = 230;   // conv out channels
constexpr int kR   = 53;    // relations
constexpr int kW   = 3;     // conv window
constexpr int kKdim = kW * kD;      // 180  (im2col K)
constexpr int kKpad = 192;          // K padded to 6 * 32
constexpr int kNpad = 240;          // OC padded to 15 * 16
constexpr int kMrows = kL + 2;      // 122 conv output positions
constexpr int kNS  = kB * kM;       // 2048 sentences

// ---- WMMA types ----
typedef __attribute__((ext_vector_type(16))) _Float16     v16h;
typedef __attribute__((ext_vector_type(8)))  float        v8f;
typedef __attribute__((ext_vector_type(4)))  unsigned int u32x4;

union Frag {
    u32x4 u[2];   // two 16-byte chunks
    v16h  h;      // 16 halves = 8 VGPRs
};

// ---- workspace layout (bytes) ----
// [0, 92160)            : BwT  f16  [240][192]  (transposed, padded conv weights)
// [92160, 140920)       : q    f32  [53][230]   (r_embed @ att_W)
// [140928, 140928+1.88M): feat f32  [2048][230]
constexpr size_t kOffBwT  = 0;
constexpr size_t kOffQ    = 92160;
constexpr size_t kOffFeat = 140928;

// =====================================================================
// Kernel 0: weight prep. BwT[n][k] = conv_w[n][k] (f16, zero padded),
//           q[r][d] = sum_c r_embed[r][c] * att_W[r][c][d]
// =====================================================================
__global__ __launch_bounds__(256) void cnn_att_prep(
    const float* __restrict__ conv_w, const float* __restrict__ r_embed,
    const float* __restrict__ att_W, _Float16* __restrict__ BwT,
    float* __restrict__ q)
{
    int idx = blockIdx.x * 256 + threadIdx.x;
    if (idx < kNpad * kKpad) {
        int nn = idx / kKpad;
        int kk = idx - nn * kKpad;
        float v = (nn < kOC && kk < kKdim) ? conv_w[nn * kKdim + kk] : 0.0f;
        BwT[idx] = (_Float16)v;
    } else {
        int j = idx - kNpad * kKpad;
        if (j < kR * kOC) {
            int r = j / kOC;
            int d = j - r * kOC;
            float acc = 0.0f;
            const float* re = r_embed + r * kOC;
            const float* aw = att_W + r * kOC * kOC + d;
            for (int c = 0; c < kOC; ++c) acc += re[c] * aw[c * kOC];
            q[j] = acc;
        }
    }
}

// =====================================================================
// Kernel 1: per-sentence conv (as im2col GEMM via WMMA) + max-pool +
//           bias + tanh  ->  feat[n][230]
// One workgroup (256 threads = 8 wave32) per sentence.
// Wave w owns N-tiles {w, w+8}: its 6 B fragments are hoisted into
// registers once (no redundant weight traffic), A fragments stream from
// LDS per M-tile, and the per-column max accumulates in registers across
// all 8 M-tiles (no cross-wave reduction needed).
// =====================================================================
__global__ __launch_bounds__(256) void cnn_att_conv(
    const int* __restrict__ word_ids, const int* __restrict__ pos1_ids,
    const int* __restrict__ pos2_ids, const float* __restrict__ w2v,
    const float* __restrict__ pos_embed, const float* __restrict__ conv_bias,
    const _Float16* __restrict__ BwT, float* __restrict__ feat_out)
{
    __shared__ __align__(16) _Float16 Alds[128 * kKpad];   // 49152 B

    const int n   = blockIdx.x;           // sentence id 0..2047
    const int tid = threadIdx.x;
    const int* wid = word_ids + n * kL;
    const int* p1  = pos1_ids + n * kL;
    const int* p2  = pos2_ids + n * kL;

    // ---- build padded im2col A [128 rows][192 cols] directly in LDS ----
    // A[t][kk] = embed(token t + kk/60 - 2, dim kk%60); zeros outside.
    for (int idx = tid; idx < 128 * kKpad; idx += 256) {
        int t  = idx / kKpad;
        int kk = idx - t * kKpad;
        float v = 0.0f;
        if (t < kMrows && kk < kKdim) {
            int kh = kk / kD;
            int d  = kk - kh * kD;
            int p  = t + kh - 2;
            if (p >= 0 && p < kL) {
                if (d < kWE)            v = w2v[wid[p] * kWE + d];
                else if (d < kWE + kPE) v = pos_embed[p1[p] * kPE + (d - kWE)];
                else                    v = pos_embed[p2[p] * kPE + (d - kWE - kPE)];
            }
        }
        Alds[idx] = (_Float16)v;
    }
    __syncthreads();

    const int wave = tid >> 5;     // 0..7
    const int lane = tid & 31;
    const int lo   = lane & 15;
    const int hi   = lane >> 4;

    for (int half = 0; half < 2; ++half) {
        const int nt = wave + half * 8;   // N-tile owned by this wave
        if (nt >= 15) break;              // wave 7 has only one tile (uniform branch)

        // ---- hoist the 6 B fragments for this N-tile into registers ----
        // B 32x16 layout: lane lo = column N; K = k*32 + 16*hi + 0..15 contiguous
        const char* brow = (const char*)(BwT + (nt * 16 + lo) * kKpad);
        Frag b[6];
#pragma unroll
        for (int k = 0; k < 6; ++k) {
            b[k].u[0] = *(const u32x4*)(brow + k * 64 + 32 * hi);
            b[k].u[1] = *(const u32x4*)(brow + k * 64 + 32 * hi + 16);
        }

        float colmax = -3.4e38f;
        for (int mt = 0; mt < 8; ++mt) {
            // A 16x32 layout: lane lo = row M; K = k*32 + 16*hi + 0..15 contiguous
            const char* arow = (const char*)&Alds[(mt * 16 + lo) * kKpad];
            Frag a[6];
#pragma unroll
            for (int k = 0; k < 6; ++k) {
                a[k].u[0] = *(const u32x4*)(arow + k * 64 + 32 * hi);
                a[k].u[1] = *(const u32x4*)(arow + k * 64 + 32 * hi + 16);
            }
            v8f c = {};
#pragma unroll
            for (int k = 0; k < 6; ++k)
                c = __builtin_amdgcn_wmma_f32_16x16x32_f16(
                        false, a[k].h, false, b[k].h, (short)0, c, false, false);
            // fold valid rows (conv positions < 122) into the running max
#pragma unroll
            for (int g = 0; g < 8; ++g) {
                int row = mt * 16 + hi * 8 + g;
                if (row < kMrows) colmax = fmaxf(colmax, c[g]);
            }
        }
        // combine the two lane halves (M = g vs M = g+8)
        colmax = fmaxf(colmax, __shfl_xor(colmax, 16));

        const int col = nt * 16 + lo;
        if (hi == 0 && col < kOC)
            feat_out[n * kOC + col] = tanhf(colmax + conv_bias[col]);
    }
}

// =====================================================================
// Kernel 2: per-bag attention + logits + log-softmax + max over r.
// One workgroup per bag.
// =====================================================================
__global__ __launch_bounds__(256) void cnn_att_pred(
    const float* __restrict__ feat, const float* __restrict__ q,
    const float* __restrict__ r_embed, const float* __restrict__ r_bias,
    float* __restrict__ out)
{
    __shared__ float fe[kM * kOC];        // 14720 B
    __shared__ float G[kM * kR];          //  3392 B  Gram: feat[m].r_embed[s]
    __shared__ float att[kR * kM];        //  3392 B  scores -> softmax weights
    __shared__ float logits[kR * 56];     // 11872 B  (padded stride)
    __shared__ float lse[kR];

    const int b   = blockIdx.x;
    const int tid = threadIdx.x;

    for (int i = tid; i < kM * kOC; i += 256) fe[i] = feat[b * kM * kOC + i];
    __syncthreads();

    // G[m][s] and raw scores att[r][m]
    for (int i = tid; i < 2 * kM * kR; i += 256) {
        if (i < kM * kR) {
            int m = i / kR, s = i - m * kR;
            float acc = 0.0f;
            const float* f = fe + m * kOC;
            const float* re = r_embed + s * kOC;
            for (int c = 0; c < kOC; ++c) acc += f[c] * re[c];
            G[m * kR + s] = acc;
        } else {
            int j = i - kM * kR;
            int r = j / kM, m = j - r * kM;
            float acc = 0.0f;
            const float* qr = q + r * kOC;
            const float* f  = fe + m * kOC;
            for (int c = 0; c < kOC; ++c) acc += qr[c] * f[c];
            att[r * kM + m] = acc;
        }
    }
    __syncthreads();

    // softmax over m for each relation r
    if (tid < kR) {
        float mx = att[tid * kM];
        for (int m = 1; m < kM; ++m) mx = fmaxf(mx, att[tid * kM + m]);
        float s = 0.0f;
        for (int m = 0; m < kM; ++m) { float e = expf(att[tid * kM + m] - mx); att[tid * kM + m] = e; s += e; }
        float inv = 1.0f / s;
        for (int m = 0; m < kM; ++m) att[tid * kM + m] *= inv;
    }
    __syncthreads();

    // logits[r][s] = 0.5 * sum_m att[r][m] * G[m][s] + r_bias[s]
    for (int i = tid; i < kR * kR; i += 256) {
        int r = i / kR, s = i - r * kR;
        float acc = 0.0f;
        for (int m = 0; m < kM; ++m) acc += att[r * kM + m] * G[m * kR + s];
        logits[r * 56 + s] = 0.5f * acc + r_bias[s];
    }
    __syncthreads();

    // log-sum-exp per relation row
    if (tid < kR) {
        float mx = logits[tid * 56];
        for (int s = 1; s < kR; ++s) mx = fmaxf(mx, logits[tid * 56 + s]);
        float ssum = 0.0f;
        for (int s = 0; s < kR; ++s) ssum += expf(logits[tid * 56 + s] - mx);
        lse[tid] = mx + logf(ssum);
    }
    __syncthreads();

    // pred[s] = max_r (logits[r][s] - lse[r])
    if (tid < kR) {
        int s = tid;
        float best = -3.4e38f;
        for (int r = 0; r < kR; ++r) best = fmaxf(best, logits[r * 56 + s] - lse[r]);
        out[b * kR + s] = best;
    }
}

// =====================================================================
extern "C" void kernel_launch(void* const* d_in, const int* in_sizes, int n_in,
                              void* d_out, int out_size, void* d_ws, size_t ws_size,
                              hipStream_t stream) {
    const int*   word_ids  = (const int*)d_in[0];
    const int*   pos1_ids  = (const int*)d_in[1];
    const int*   pos2_ids  = (const int*)d_in[2];
    const float* w2v       = (const float*)d_in[3];
    const float* pos_embed = (const float*)d_in[4];
    const float* conv_w    = (const float*)d_in[5];
    const float* conv_bias = (const float*)d_in[6];
    const float* r_embed   = (const float*)d_in[7];
    const float* r_bias    = (const float*)d_in[8];
    const float* att_W     = (const float*)d_in[9];

    char* ws = (char*)d_ws;
    _Float16* BwT  = (_Float16*)(ws + kOffBwT);
    float*    q    = (float*)(ws + kOffQ);
    float*    feat = (float*)(ws + kOffFeat);

    int prep_tasks = kNpad * kKpad + kR * kOC;          // 58270
    cnn_att_prep<<<(prep_tasks + 255) / 256, 256, 0, stream>>>(
        conv_w, r_embed, att_W, BwT, q);

    cnn_att_conv<<<kNS, 256, 0, stream>>>(
        word_ids, pos1_ids, pos2_ids, w2v, pos_embed, conv_bias, BwT, feat);

    cnn_att_pred<<<kB, 256, 0, stream>>>(
        feat, q, r_embed, r_bias, (float*)d_out);
}
